// NTXentLoss_21887153340916
// MI455X (gfx1250) — matrix operand order repaired and checked by reading, not compiled, not run on hardware
//
#include <hip/hip_runtime.h>
#include <hip/hip_bf16.h>

typedef __attribute__((ext_vector_type(16))) __bf16 v16bf;
typedef __attribute__((ext_vector_type(8)))  float  v8f;

#define B_TOTAL 8192
#define D_DIM   256
#define INV_T   10.0f
// 10 * log2(e)
#define KSCALE  14.426950408889634f
#define LN2     0.6931471805599453f

#define COLS_PER_BLK 32
#define LDS_STRIDE   272   // 256 + 16 pad: spreads DS banks for b128 reads
#define PANEL_ELEMS  (COLS_PER_BLK * LDS_STRIDE)   // ushorts per panel
// dynamic LDS: [hi buf0][hi buf1][lo buf0][lo buf1]
#define LDS_BYTES    (4 * PANEL_ELEMS * 2)

// 32-byte POD for bit_cast into v16bf
struct Bytes32 { uint4 a, b; };

__device__ __forceinline__ v16bf load_frag32B(const unsigned short* p) {
    Bytes32 t;
    t.a = *(const uint4*)(p);
    t.b = *(const uint4*)(p + 8);
    return __builtin_bit_cast(v16bf, t);
}

// ---------------------------------------------------------------------------
// Kernel 1: L2-normalize rows of latent, emit bf16 hi/lo split panels.
// ---------------------------------------------------------------------------
__global__ __launch_bounds__(256) void normalize_split_kernel(
    const float* __restrict__ latent,
    unsigned short* __restrict__ zh,
    unsigned short* __restrict__ zl,
    float* __restrict__ sums)
{
    if (blockIdx.x == 0 && threadIdx.x == 0) { sums[0] = 0.0f; sums[1] = 0.0f; }

    const int row  = blockIdx.x * 8 + (threadIdx.x >> 5);
    const int lane = threadIdx.x & 31;

    const float4* src = (const float4*)(latent + row * D_DIM + lane * 8);
    float4 v0 = src[0];
    float4 v1 = src[1];

    float ss = v0.x*v0.x + v0.y*v0.y + v0.z*v0.z + v0.w*v0.w
             + v1.x*v1.x + v1.y*v1.y + v1.z*v1.z + v1.w*v1.w;
    #pragma unroll
    for (int m = 16; m >= 1; m >>= 1) ss += __shfl_xor(ss, m, 32);

    const float rn = rsqrtf(ss);

    float z[8] = { v0.x*rn, v0.y*rn, v0.z*rn, v0.w*rn,
                   v1.x*rn, v1.y*rn, v1.z*rn, v1.w*rn };

    union { unsigned short u[8]; uint4 v; } ph, pl;
    #pragma unroll
    for (int j = 0; j < 8; ++j) {
        __bf16 hb = (__bf16)z[j];
        float  hf = (float)hb;
        __bf16 lb = (__bf16)(z[j] - hf);
        ph.u[j] = __builtin_bit_cast(unsigned short, hb);
        pl.u[j] = __builtin_bit_cast(unsigned short, lb);
    }
    *(uint4*)(zh + row * D_DIM + lane * 8) = ph.v;
    *(uint4*)(zl + row * D_DIM + lane * 8) = pl.v;
}

// ---------------------------------------------------------------------------
// Kernel 2: fused sim-GEMM + masked softmax statistics, software-pipelined:
// global loads for panel i+1 are issued before the WMMA phase of panel i,
// written to the alternate LDS buffer after compute (1 barrier / iteration).
// ---------------------------------------------------------------------------
__global__ __launch_bounds__(256) void supcon_main_kernel(
    const unsigned short* __restrict__ zh,
    const unsigned short* __restrict__ zl,
    const int* __restrict__ labels,
    float* __restrict__ sums)
{
    extern __shared__ __align__(16) unsigned short smem[];

    const int tid  = threadIdx.x;
    const int wave = tid >> 5;
    const int lane = tid & 31;
    const int n    = lane & 15;   // column-within-tile / A-row selector
    const int g    = lane >> 4;   // half-wave group

    const int rowbase = blockIdx.x * 128 + wave * 16;
    const int arow    = rowbase + n;

    // Staging role: 8 threads per column, 32 elems (64 B) per component each.
    const int crow = tid >> 3;
    const int seg  = tid & 7;

    // Resident A fragments: 16 rows x K=256, hi + lo (128 VGPRs).
    v16bf a_hi[8], a_lo[8];
    #pragma unroll
    for (int kc = 0; kc < 8; ++kc) {
        const int k0 = kc * 32 + g * 16;
        Bytes32 t;
        t.a = *(const uint4*)(zh + arow * D_DIM + k0);
        t.b = *(const uint4*)(zh + arow * D_DIM + k0 + 8);
        a_hi[kc] = __builtin_bit_cast(v16bf, t);
        t.a = *(const uint4*)(zl + arow * D_DIM + k0);
        t.b = *(const uint4*)(zl + arow * D_DIM + k0 + 8);
        a_lo[kc] = __builtin_bit_cast(v16bf, t);
    }

    int lrow[8];
    #pragma unroll
    for (int r = 0; r < 8; ++r) lrow[r] = labels[rowbase + r + g * 8];

    // Per-lane accumulators (lane n covers all columns == n (mod 16)).
    float l_acc[8], ps[8], pc[8];
    #pragma unroll
    for (int r = 0; r < 8; ++r) { l_acc[r] = 0.f; ps[r] = 0.f; pc[r] = 0.f; }

    uint4 rh[4], rl[4];   // staging registers for the next panel

    // Prologue: stage panel 0 into buffer 0.
    {
        const uint4* gh = (const uint4*)(zh + crow * D_DIM + seg * 32);
        const uint4* gl = (const uint4*)(zl + crow * D_DIM + seg * 32);
        uint4* dh = (uint4*)(smem + 0 * PANEL_ELEMS + crow * LDS_STRIDE + seg * 32);
        uint4* dl = (uint4*)(smem + 2 * PANEL_ELEMS + crow * LDS_STRIDE + seg * 32);
        #pragma unroll
        for (int j = 0; j < 4; ++j) { dh[j] = gh[j]; dl[j] = gl[j]; }
    }
    __syncthreads();

    const int NBLK = B_TOTAL / COLS_PER_BLK;
    int cur = 0;
    for (int ib = 0; ib < NBLK; ++ib) {
        const int cb = ib * COLS_PER_BLK;
        const bool has_next = (ib + 1) < NBLK;

        // Issue global loads for the NEXT panel now; latency hides under WMMA.
        if (has_next) {
            const int cb2 = cb + COLS_PER_BLK;
            const uint4* gh = (const uint4*)(zh + (cb2 + crow) * D_DIM + seg * 32);
            const uint4* gl = (const uint4*)(zl + (cb2 + crow) * D_DIM + seg * 32);
            #pragma unroll
            for (int j = 0; j < 4; ++j) { rh[j] = gh[j]; rl[j] = gl[j]; }
        }

        // Compute the two 16x16 tiles of this panel.
        const unsigned short* baseH = smem + cur * PANEL_ELEMS;
        const unsigned short* baseL = smem + (2 + cur) * PANEL_ELEMS;
        #pragma unroll
        for (int ct = 0; ct < 2; ++ct) {
            const int colLocal = ct * 16 + n;
            const int col      = cb + colLocal;
            const unsigned short* bh = baseH + colLocal * LDS_STRIDE + g * 16;
            const unsigned short* bl = baseL + colLocal * LDS_STRIDE + g * 16;

            v8f c = {};
            #pragma unroll
            for (int kc = 0; kc < 8; ++kc) {
                v16bf Bh = load_frag32B(bh + kc * 32);
                v16bf Bl = load_frag32B(bl + kc * 32);
                c = __builtin_amdgcn_wmma_f32_16x16x32_bf16(
                        false, a_hi[kc], false, Bh, (short)0, c, false, false);
                c = __builtin_amdgcn_wmma_f32_16x16x32_bf16(
                        false, a_hi[kc], false, Bl, (short)0, c, false, false);
                c = __builtin_amdgcn_wmma_f32_16x16x32_bf16(
                        false, a_lo[kc], false, Bh, (short)0, c, false, false);
            }

            const int lcol = labels[col];
            #pragma unroll
            for (int r = 0; r < 8; ++r) {
                const float dot = c[r];
                const float s   = dot * INV_T;
                const int   row = rowbase + r + g * 8;
                const bool diag = (row == col);
                // exp(s - 10) == exp2(dot*10*log2e - 10*log2e); cos-sim <= 1
                // so no overflow — fixed shift replaces online-max tracking.
                float e = __builtin_amdgcn_exp2f(fmaf(dot, KSCALE, -KSCALE));
                l_acc[r] += diag ? 0.0f : e;
                const bool pos = (lcol == lrow[r]) && !diag;
                ps[r] += pos ? s : 0.0f;
                pc[r] += pos ? 1.0f : 0.0f;
            }
        }

        // Drain staged data into the alternate buffer, then one barrier.
        if (has_next) {
            unsigned short* nH = smem + (cur ^ 1) * PANEL_ELEMS;
            unsigned short* nL = smem + (2 + (cur ^ 1)) * PANEL_ELEMS;
            uint4* dh = (uint4*)(nH + crow * LDS_STRIDE + seg * 32);
            uint4* dl = (uint4*)(nL + crow * LDS_STRIDE + seg * 32);
            #pragma unroll
            for (int j = 0; j < 4; ++j) { dh[j] = rh[j]; dl[j] = rl[j]; }
            __syncthreads();
            cur ^= 1;
        }
    }

    // Cross-lane (width-16) reduction, per-row loss, two atomics per half-wave.
    float lsum = 0.0f, csum = 0.0f;
    #pragma unroll
    for (int r = 0; r < 8; ++r) {
        float l = l_acc[r], p = ps[r], q = pc[r];
        #pragma unroll
        for (int m = 1; m < 16; m <<= 1) {
            l += __shfl_xor(l, m, 16);
            p += __shfl_xor(p, m, 16);
            q += __shfl_xor(q, m, 16);
        }
        if (q > 0.0f) {
            const float lse = INV_T + __builtin_amdgcn_logf(l) * LN2;
            lsum += lse - p / q;   // loss_i = -(mean_pos - lse)
            csum += 1.0f;
        }
    }
    if (n == 0) {
        atomicAdd(&sums[0], lsum);
        atomicAdd(&sums[1], csum);
    }
}

// ---------------------------------------------------------------------------
// Kernel 3: finalize scalar loss.
// ---------------------------------------------------------------------------
__global__ void finalize_kernel(const float* __restrict__ sums,
                                float* __restrict__ out)
{
    if (threadIdx.x == 0 && blockIdx.x == 0) {
        const float cnt = sums[1];
        out[0] = (cnt > 0.0f) ? (sums[0] / cnt) : 0.0f;
    }
}

extern "C" void kernel_launch(void* const* d_in, const int* in_sizes, int n_in,
                              void* d_out, int out_size, void* d_ws, size_t ws_size,
                              hipStream_t stream) {
    (void)in_sizes; (void)n_in; (void)out_size; (void)ws_size;

    const float* latent = (const float*)d_in[0];
    const int*   labels = (const int*)d_in[1];
    float*       out    = (float*)d_out;

    // Workspace layout: [zh bf16 8192x256][zl bf16 8192x256][2 floats]
    unsigned short* zh = (unsigned short*)d_ws;
    unsigned short* zl = zh + (size_t)B_TOTAL * D_DIM;
    float* sums = (float*)((char*)d_ws + (size_t)2 * B_TOTAL * D_DIM * 2);

    normalize_split_kernel<<<B_TOTAL / 8, 256, 0, stream>>>(latent, zh, zl, sums);
    supcon_main_kernel<<<B_TOTAL / 128, 256, LDS_BYTES, stream>>>(zh, zl, labels, sums);
    finalize_kernel<<<1, 32, 0, stream>>>(sums, out);
}
